// TemporalAttention_39822936768786
// MI455X (gfx1250) — matrix operand order repaired and checked by reading, not compiled, not run on hardware
//
#include <hip/hip_runtime.h>
#include <math.h>

// ---------------------------------------------------------------------------
// CDNA5 (gfx1250) bf16 WMMA attention pipeline, async-LDS staged.
// Compute-bound (~275 GFLOP vs ~200MB HBM) -> all GEMMs via
// v_wmma_f32_16x16x32_bf16, fed by GLOBAL_LOAD_ASYNC_TO_LDS_B128 double
// buffering (ASYNCcnt) and ds_load_b128 fragment reads.
// ---------------------------------------------------------------------------

typedef __attribute__((ext_vector_type(16))) __bf16 bf16x16;
typedef __attribute__((ext_vector_type(8)))  __bf16 bf16x8;
typedef __attribute__((ext_vector_type(8)))  float  f32x8;

union AFragU { bf16x16 v; bf16x8 h[2]; };

__device__ __forceinline__ unsigned short f32_to_bf16(float f) {
    union { float f; unsigned u; } cv; cv.f = f;
    unsigned u = cv.u;
    u += 0x7FFFu + ((u >> 16) & 1u);   // round-to-nearest-even
    return (unsigned short)(u >> 16);
}
__device__ __forceinline__ float bf16_to_f32(unsigned short h) {
    union { unsigned u; float f; } cv; cv.u = ((unsigned)h) << 16;
    return cv.f;
}

// 32-bit LDS offset of a __shared__ object (generic AS3->flat cast keeps the
// LDS byte offset in the low 32 bits).
__device__ __forceinline__ unsigned lds_off_u32(const void* p) {
    return (unsigned)(unsigned long long)p;
}

// One 16B async copy: global -> LDS, tracked by ASYNCcnt. Per-lane addresses,
// so one wave instruction moves 512B.
__device__ __forceinline__ void async_copy_b128(unsigned lds_off,
                                                const unsigned short* gsrc) {
    asm volatile("global_load_async_to_lds_b128 %0, %1, off"
                 :: "v"(lds_off), "v"(gsrc)
                 : "memory");
}
__device__ __forceinline__ void wait_async_all() {
    asm volatile("s_wait_asynccnt 0" ::: "memory");
}

// ---------------------------------------------------------------------------
// Tiling: block = 256 threads = 8 wave32s (2 x 4). Wave tile 32x64 = 2x4
// WMMA tiles; block tile 64x256, K-step 32. LDS rows padded to 40 shorts
// (80B = 20 banks; 16 lanes * stride-20-banks -> conflict-free b128 reads).
// ---------------------------------------------------------------------------
#define BLK_M   64
#define BLK_N   256
#define KSTEP   32
#define LDSW    40   // shorts per LDS row (32 data + 8 pad)

// A fragment from LDS: 16x32 (MxK). Lane L: row = base+(L&15),
// chunks at K = kh..kh+7 and kh+16..kh+23, kh = (L>>4)*8.
__device__ __forceinline__ bf16x16
lds_a_frag(const unsigned short* base, int row0, int lane) {
    const unsigned short* p = base + (row0 + (lane & 15)) * LDSW + ((lane >> 4) << 3);
    AFragU f;
    f.h[0] = *(const bf16x8*)(p);
    f.h[1] = *(const bf16x8*)(p + 16);
    return f.v;
}
// B fragment from LDS: 32x16 (KxN) sourced from row-major [N x K] tile.
// Lane L: col = base+(L&15), K = kh..kh+7 and kh+16..kh+23, kh = (L>>4)*8 (2 x b128).
__device__ __forceinline__ bf16x16
lds_b_frag(const unsigned short* base, int col0, int lane) {
    const unsigned short* p = base + (col0 + (lane & 15)) * LDSW + ((lane >> 4) << 3);
    AFragU f;
    f.h[0] = *(const bf16x8*)(p);
    f.h[1] = *(const bf16x8*)(p + 16);
    return f.v;
}

// ---------------------------------------------------------------------------
// Generic bf16 WMMA GEMM: C = A(MxK) * Bm(NxK)^T with mode-specific epilogue.
// mode 0: out_bf[grow*N+c]            = bf16(acc + bias[c])                 (Q)
// mode 1: out_bf[grow*N+c]            = bf16(a*s*(acc+bias[c]) + ab[c*T+s]) (K')
// mode 2: out_bf[b*N*T + c*T + srow]  = bf16(acc + bias[c])                 (V^T)
// mode 3: out_bf[grow*N+c]            = bf16(sigmoid(acc + bias[c]))        (gate)
// mode 4: out_f [grow*N+c]            = acc                                 (scores)
// mode 5: out_f [grow*N+c]            = acc * gate[grow*N+c]                (P.V)
// Requires M%64==0, N%256==0, K%32==0 (true for all shapes here).
// ---------------------------------------------------------------------------
__global__ __launch_bounds__(256)
void wmma_gemm_kernel(const unsigned short* __restrict__ A,
                      const unsigned short* __restrict__ Bm,
                      int M, int N, int K, int Tt,
                      long long aBatch, long long bBatch,
                      int mode, float scale,
                      const float* __restrict__ bias,
                      const float* __restrict__ attn_bias,
                      const float* __restrict__ alphap,
                      const unsigned short* __restrict__ gate,
                      unsigned short* __restrict__ out_bf,
                      float* __restrict__ out_f) {
    __shared__ __align__(16) unsigned short sA[2][BLK_M * LDSW];
    __shared__ __align__(16) unsigned short sB[2][BLK_N * LDSW];

    const int tid  = threadIdx.x;
    const int lane = tid & 31;
    const int wave = tid >> 5;
    const int wm = wave & 1;        // 0..1  -> 32-row slice
    const int wn = wave >> 1;       // 0..3  -> 64-col slice

    const int rowBlk = blockIdx.y * BLK_M;
    const int colBlk = blockIdx.x * BLK_N;
    const int row0 = rowBlk + wm * 32;
    const int col0 = colBlk + wn * 64;

    const unsigned short* Az = A + (size_t)blockIdx.z * (size_t)aBatch;
    const unsigned short* Bz = Bm + (size_t)blockIdx.z * (size_t)bBatch;
    const long long zrow = (long long)blockIdx.z * M;

    // Per-thread async-copy assignments (16B chunks; 4 chunks per 32-K row).
    const int ar = tid >> 2;          // A row 0..63
    const int ac = (tid & 3) << 3;    // A k-chunk offset in shorts

    f32x8 acc[2][4] = {};

    // ---- stage 0 prologue ----
    {
        const int k0 = 0;
        async_copy_b128(lds_off_u32(&sA[0][ar * LDSW + ac]),
                        Az + (size_t)(rowBlk + ar) * K + (k0 + ac));
        #pragma unroll
        for (int i = 0; i < 4; ++i) {
            const int chunk = (i << 8) + tid;       // 0..1023
            const int br = chunk >> 2;              // B row 0..255
            const int bc = (chunk & 3) << 3;
            async_copy_b128(lds_off_u32(&sB[0][br * LDSW + bc]),
                            Bz + (size_t)(colBlk + br) * K + (k0 + bc));
        }
    }
    wait_async_all();
    __syncthreads();

    int buf = 0;
    for (int k0 = 0; k0 < K; k0 += KSTEP) {
        // ---- issue next stage into the other buffer ----
        if (k0 + KSTEP < K) {
            const int kn = k0 + KSTEP;
            const int nb = buf ^ 1;
            async_copy_b128(lds_off_u32(&sA[nb][ar * LDSW + ac]),
                            Az + (size_t)(rowBlk + ar) * K + (kn + ac));
            #pragma unroll
            for (int i = 0; i < 4; ++i) {
                const int chunk = (i << 8) + tid;
                const int br = chunk >> 2;
                const int bc = (chunk & 3) << 3;
                async_copy_b128(lds_off_u32(&sB[nb][br * LDSW + bc]),
                                Bz + (size_t)(colBlk + br) * K + (kn + bc));
            }
        }

        // ---- compute current stage from LDS ----
        bf16x16 a0 = lds_a_frag(sA[buf], wm * 32,      lane);
        bf16x16 a1 = lds_a_frag(sA[buf], wm * 32 + 16, lane);
        bf16x16 b0 = lds_b_frag(sB[buf], wn * 64,      lane);
        bf16x16 b1 = lds_b_frag(sB[buf], wn * 64 + 16, lane);
        bf16x16 b2 = lds_b_frag(sB[buf], wn * 64 + 32, lane);
        bf16x16 b3 = lds_b_frag(sB[buf], wn * 64 + 48, lane);

        acc[0][0] = __builtin_amdgcn_wmma_f32_16x16x32_bf16(false, a0, false, b0, (short)0, acc[0][0], false, false);
        acc[0][1] = __builtin_amdgcn_wmma_f32_16x16x32_bf16(false, a0, false, b1, (short)0, acc[0][1], false, false);
        acc[0][2] = __builtin_amdgcn_wmma_f32_16x16x32_bf16(false, a0, false, b2, (short)0, acc[0][2], false, false);
        acc[0][3] = __builtin_amdgcn_wmma_f32_16x16x32_bf16(false, a0, false, b3, (short)0, acc[0][3], false, false);
        acc[1][0] = __builtin_amdgcn_wmma_f32_16x16x32_bf16(false, a1, false, b0, (short)0, acc[1][0], false, false);
        acc[1][1] = __builtin_amdgcn_wmma_f32_16x16x32_bf16(false, a1, false, b1, (short)0, acc[1][1], false, false);
        acc[1][2] = __builtin_amdgcn_wmma_f32_16x16x32_bf16(false, a1, false, b2, (short)0, acc[1][2], false, false);
        acc[1][3] = __builtin_amdgcn_wmma_f32_16x16x32_bf16(false, a1, false, b3, (short)0, acc[1][3], false, false);

        // next-stage copies complete + all waves done before buffer swap
        wait_async_all();
        __syncthreads();
        buf ^= 1;
    }

    // C/D layout: lane L holds col n = L&15; rows m = (L>>4)*8 + i for VGPR i.
    const float am = (mode == 1) ? (alphap[0] * scale) : 0.0f;

    for (int mi = 0; mi < 2; ++mi) {
        for (int ni = 0; ni < 4; ++ni) {
            const int rbase = row0 + mi * 16 + ((lane >> 4) << 3);
            const int c     = col0 + ni * 16 + (lane & 15);
            const f32x8 a = acc[mi][ni];
            const float bv = (mode <= 3) ? bias[c] : 0.0f;

            if (mode == 0) {
                #pragma unroll
                for (int i = 0; i < 8; ++i) {
                    long long grow = zrow + rbase + i;
                    out_bf[(size_t)grow * N + c] = f32_to_bf16(a[i] + bv);
                }
            } else if (mode == 1) {
                #pragma unroll
                for (int i = 0; i < 8; ++i) {
                    long long grow = zrow + rbase + i;
                    int srow = (int)(grow % Tt);
                    float v = am * (a[i] + bv) + attn_bias[(size_t)c * Tt + srow];
                    out_bf[(size_t)grow * N + c] = f32_to_bf16(v);
                }
            } else if (mode == 2) {
                #pragma unroll
                for (int i = 0; i < 8; ++i) {
                    long long grow = zrow + rbase + i;
                    long long b = grow / Tt;
                    int srow = (int)(grow % Tt);
                    out_bf[(size_t)b * N * Tt + (size_t)c * Tt + srow] = f32_to_bf16(a[i] + bv);
                }
            } else if (mode == 3) {
                #pragma unroll
                for (int i = 0; i < 8; ++i) {
                    long long grow = zrow + rbase + i;
                    float v = a[i] + bv;
                    float g = 1.0f / (1.0f + __expf(-v));
                    out_bf[(size_t)grow * N + c] = f32_to_bf16(g);
                }
            } else if (mode == 4) {
                #pragma unroll
                for (int i = 0; i < 8; ++i) {
                    long long grow = zrow + rbase + i;
                    out_f[(size_t)grow * N + c] = a[i];
                }
            } else { // mode 5
                #pragma unroll
                for (int i = 0; i < 8; ++i) {
                    long long grow = zrow + rbase + i;
                    float g = bf16_to_f32(gate[(size_t)grow * N + c]);
                    out_f[(size_t)grow * N + c] = a[i] * g;
                }
            }
        }
    }
}

// ---------------------------------------------------------------------------
// Row softmax over T elements, in place on f32 scores, also emits bf16 copy.
// One 256-thread block (8 wave32s) per row; wave32 shuffle + LDS reduce.
// ---------------------------------------------------------------------------
__global__ __launch_bounds__(256)
void softmax_rows_kernel(float* __restrict__ S,
                         unsigned short* __restrict__ Pb,
                         int Tlen) {
    __shared__ float red[8];
    const size_t row = blockIdx.x;
    float* s = S + row * (size_t)Tlen;
    const int tid = threadIdx.x;
    const int per = Tlen >> 8;           // elements per thread (T/256)

    float v[16];
    float m = -3.402823466e38f;
    for (int i = 0; i < per; ++i) {
        v[i] = s[tid + (i << 8)];
        m = fmaxf(m, v[i]);
    }
    for (int off = 16; off > 0; off >>= 1)
        m = fmaxf(m, __shfl_xor(m, off, 32));
    if ((tid & 31) == 0) red[tid >> 5] = m;
    __syncthreads();
    float mAll = red[0];
    #pragma unroll
    for (int i = 1; i < 8; ++i) mAll = fmaxf(mAll, red[i]);
    __syncthreads();

    float acc = 0.0f;
    for (int i = 0; i < per; ++i) {
        v[i] = __expf(v[i] - mAll);
        acc += v[i];
    }
    for (int off = 16; off > 0; off >>= 1)
        acc += __shfl_xor(acc, off, 32);
    if ((tid & 31) == 0) red[tid >> 5] = acc;
    __syncthreads();
    float tot = 0.0f;
    #pragma unroll
    for (int i = 0; i < 8; ++i) tot += red[i];
    const float inv = 1.0f / tot;

    unsigned short* p = Pb + row * (size_t)Tlen;
    for (int i = 0; i < per; ++i) {
        const float w = v[i] * inv;
        const int idx = tid + (i << 8);
        s[idx] = w;                       // f32 attn_weights (d_out)
        p[idx] = f32_to_bf16(w);          // bf16 P for the P.V WMMA GEMM
    }
}

// ---------------------------------------------------------------------------
// f32 -> bf16 bulk convert (grid-stride).
// ---------------------------------------------------------------------------
__global__ __launch_bounds__(256)
void cvt_f32_bf16_kernel(const float* __restrict__ in,
                         unsigned short* __restrict__ out,
                         long long n) {
    long long i = (long long)blockIdx.x * blockDim.x + threadIdx.x;
    const long long stride = (long long)gridDim.x * blockDim.x;
    for (; i < n; i += stride) out[i] = f32_to_bf16(in[i]);
}

// ---------------------------------------------------------------------------
extern "C" void kernel_launch(void* const* d_in, const int* in_sizes, int n_in,
                              void* d_out, int out_size, void* d_ws, size_t ws_size,
                              hipStream_t stream) {
    const float* x         = (const float*)d_in[0];
    const float* Wq        = (const float*)d_in[1];
    const float* bq        = (const float*)d_in[2];
    const float* Wk        = (const float*)d_in[3];
    const float* bk        = (const float*)d_in[4];
    const float* Wv        = (const float*)d_in[5];
    const float* bv        = (const float*)d_in[6];
    const float* Wg        = (const float*)d_in[7];
    const float* bg        = (const float*)d_in[8];
    const float* alpha     = (const float*)d_in[9];
    const float* attn_bias = (const float*)d_in[10];

    const int D = in_sizes[2];                       // 1024
    const int T = in_sizes[10] / D;                  // 2048
    const long long BT = (long long)in_sizes[0] / D; // 16384
    const int Bn = (int)(BT / T);                    // 8
    const float scale = 1.0f / sqrtf((float)D);

    // -------- workspace carve-up (bf16 buffers) --------
    unsigned char* w = (unsigned char*)d_ws;
    auto carve = [&](size_t bytes) {
        void* p = (void*)w;
        w += (bytes + 255) & ~(size_t)255;
        return p;
    };
    unsigned short* xb  = (unsigned short*)carve((size_t)BT * D * 2);
    unsigned short* Wqb = (unsigned short*)carve((size_t)D * D * 2);
    unsigned short* Wkb = (unsigned short*)carve((size_t)D * D * 2);
    unsigned short* Wvb = (unsigned short*)carve((size_t)D * D * 2);
    unsigned short* Wgb = (unsigned short*)carve((size_t)D * D * 2);
    unsigned short* Qb  = (unsigned short*)carve((size_t)BT * D * 2);
    unsigned short* Kpb = (unsigned short*)carve((size_t)BT * D * 2);  // K' (alpha/scale/bias folded)
    unsigned short* VTb = (unsigned short*)carve((size_t)BT * D * 2);  // V^T: [B][D][T]
    unsigned short* Gb  = (unsigned short*)carve((size_t)BT * D * 2);  // sigmoid gate
    unsigned short* Pb  = (unsigned short*)carve((size_t)BT * T * 2);  // attn weights bf16

    float* outAttn = (float*)d_out;                  // [B,T,D]
    float* outW    = (float*)d_out + (size_t)BT * D; // [B,T,T]

    // -------- 1) convert x and weights to bf16 --------
    cvt_f32_bf16_kernel<<<2048, 256, 0, stream>>>(x,  xb,  BT * (long long)D);
    cvt_f32_bf16_kernel<<<512,  256, 0, stream>>>(Wq, Wqb, (long long)D * D);
    cvt_f32_bf16_kernel<<<512,  256, 0, stream>>>(Wk, Wkb, (long long)D * D);
    cvt_f32_bf16_kernel<<<512,  256, 0, stream>>>(Wv, Wvb, (long long)D * D);
    cvt_f32_bf16_kernel<<<512,  256, 0, stream>>>(Wg, Wgb, (long long)D * D);

    const dim3 blk(256);
    const dim3 gProj(D / 256, (unsigned)(BT / 64), 1);

    // -------- 2) projections (Q, K', V^T) --------
    wmma_gemm_kernel<<<gProj, blk, 0, stream>>>(xb, Wqb, (int)BT, D, D, T, 0, 0,
        /*mode=*/0, scale, bq, nullptr, nullptr, nullptr, Qb, nullptr);
    wmma_gemm_kernel<<<gProj, blk, 0, stream>>>(xb, Wkb, (int)BT, D, D, T, 0, 0,
        /*mode=*/1, scale, bk, attn_bias, alpha, nullptr, Kpb, nullptr);
    wmma_gemm_kernel<<<gProj, blk, 0, stream>>>(xb, Wvb, (int)BT, D, D, T, 0, 0,
        /*mode=*/2, scale, bv, nullptr, nullptr, nullptr, VTb, nullptr);

    // -------- 3) gate = sigmoid(Q Wg^T + bg) --------
    wmma_gemm_kernel<<<gProj, blk, 0, stream>>>(Qb, Wgb, (int)BT, D, D, T, 0, 0,
        /*mode=*/3, scale, bg, nullptr, nullptr, nullptr, Gb, nullptr);

    // -------- 4) raw scores = Q K'^T (into attn_weights region of d_out) --------
    const dim3 gScore(T / 256, T / 64, Bn);
    wmma_gemm_kernel<<<gScore, blk, 0, stream>>>(Qb, Kpb, T, T, D, T,
        (long long)T * D, (long long)T * D,
        /*mode=*/4, scale, nullptr, nullptr, nullptr, nullptr, nullptr, outW);

    // -------- 5) softmax rows (in place) + bf16 P --------
    softmax_rows_kernel<<<(unsigned)BT, blk, 0, stream>>>(outW, Pb, T);

    // -------- 6) out = (P V) * gate --------
    const dim3 gPV(D / 256, T / 64, Bn);
    wmma_gemm_kernel<<<gPV, blk, 0, stream>>>(Pb, VTb, T, D, T, T,
        (long long)T * T, (long long)D * T,
        /*mode=*/5, scale, nullptr, nullptr, nullptr, Gb, nullptr, outAttn);
}
